// MistralAttention_63617055588932
// MI455X (gfx1250) — matrix operand order repaired and checked by reading, not compiled, not run on hardware
//
#include <hip/hip_runtime.h>
#include <hip/hip_bf16.h>
#include <math.h>

#define NUM_HEADS_C 32
#define NUM_KV_C    8
#define HEAD_DIM_C  128
#define HIDDEN_C    4096
#define KVDIM_C     (NUM_KV_C * HEAD_DIM_C)   /* 1024 */
#define BATCH_C     2
#define SEQ_C       2048
#define TOKENS_C    (BATCH_C * SEQ_C)         /* 4096 */

typedef __bf16 bf16_t;
typedef __attribute__((ext_vector_type(16))) __bf16 v16bf;
typedef __attribute__((ext_vector_type(8)))  __bf16 bf16x8;
typedef __attribute__((ext_vector_type(8)))  float  v8f;
typedef __attribute__((ext_vector_type(4)))  float  f32x4;
typedef __attribute__((ext_vector_type(4)))  int    v4i;

// ---------------------------------------------------------------------------
// gfx1250 async global->LDS copy (ASYNCcnt-tracked). Guarded: falls back to
// load + ds_store if the builtin is not exposed by this toolchain.
// Probe-derived signature: (v4i addrspace(1)*, v4i addrspace(3)*, imm, imm)
// ---------------------------------------------------------------------------
#if defined(__has_builtin)
#  if __has_builtin(__builtin_amdgcn_global_load_async_to_lds_b128) && \
      __has_builtin(__builtin_amdgcn_s_wait_asynccnt)
#    define HAVE_ASYNC_LDS 1
#  endif
#endif
#ifndef HAVE_ASYNC_LDS
#  define HAVE_ASYNC_LDS 0
#endif

__device__ __forceinline__ void async_cp16(bf16_t* lds_dst, const bf16_t* gsrc) {
#if HAVE_ASYNC_LDS
  __builtin_amdgcn_global_load_async_to_lds_b128(
      (__attribute__((address_space(1))) v4i*)gsrc,
      (__attribute__((address_space(3))) v4i*)lds_dst,
      /*offset=*/0, /*cpol=*/0);
#else
  *(bf16x8*)lds_dst = *(const bf16x8*)gsrc;
#endif
}

__device__ __forceinline__ void async_join() {
#if HAVE_ASYNC_LDS
  __builtin_amdgcn_s_wait_asynccnt(0);
#endif
}

// ---------------------------------------------------------------------------
// WMMA wrapper: D = A(16x32 bf16) * B(32x16 bf16) + C(16x16 f32)
// ---------------------------------------------------------------------------
__device__ __forceinline__ v8f wmma_bf16(v16bf a, v16bf b, v8f c) {
  return __builtin_amdgcn_wmma_f32_16x16x32_bf16(
      /*neg_a=*/false, a, /*neg_b=*/false, b,
      /*c_mod=*/(short)0, c, /*reuse_a=*/false, /*reuse_b=*/false);
}

// Copy 8 contiguous bf16 from memory/LDS into fragment slots [o, o+8)
__device__ __forceinline__ void cp8(v16bf& f, int o, const bf16_t* p) {
  bf16x8 x = *(const bf16x8*)p;
#pragma unroll
  for (int i = 0; i < 8; ++i) f[o + i] = x[i];
}

// Load 8 contiguous f32 from memory, convert to bf16 into slots [o, o+8)
__device__ __forceinline__ void cvt8(v16bf& f, int o, const float* p) {
  f32x4 x = *(const f32x4*)p;
  f32x4 y = *(const f32x4*)(p + 4);
  f[o+0]=(bf16_t)x.x; f[o+1]=(bf16_t)x.y; f[o+2]=(bf16_t)x.z; f[o+3]=(bf16_t)x.w;
  f[o+4]=(bf16_t)y.x; f[o+5]=(bf16_t)y.y; f[o+6]=(bf16_t)y.z; f[o+7]=(bf16_t)y.w;
}

__device__ __forceinline__ v8f vzero8() {
  v8f z = {0.f,0.f,0.f,0.f,0.f,0.f,0.f,0.f};
  return z;
}

// ---------------------------------------------------------------------------
// GEMM: C[m,n] = sum_k A[m,k] * W[n,k]   (x @ W^T, both row-major, K contig)
// A: fp32 (ABF16=false) or bf16 (ABF16=true). W: fp32, converted per-fragment.
// mode 0: store f32 row-major (ldc=N)
// mode 1: store bf16 row-major (ldc=N)
// mode 2: store bf16 transposed for V: Vt[((b*KVH + kvh)*128 + d)*SEQ + s]
// Block: 256 threads = 8 waves; wave tile 64x64; block tile 128(M) x 256(N).
// ---------------------------------------------------------------------------
template <bool ABF16>
__global__ __launch_bounds__(256) void gemm_xwT(const void* __restrict__ Ap,
                                                const float* __restrict__ W,
                                                int K, int N,
                                                float* __restrict__ outF,
                                                bf16_t* __restrict__ outB,
                                                int mode) {
  const int lane = threadIdx.x & 31;
  const int wid  = threadIdx.x >> 5;
  const int lo = lane & 15, hi = lane >> 4;
  const int m0 = blockIdx.x * 128 + (wid & 1) * 64;
  const int n0 = blockIdx.y * 256 + (wid >> 1) * 64;

  v8f acc[4][4];
#pragma unroll
  for (int i = 0; i < 4; ++i)
#pragma unroll
    for (int j = 0; j < 4; ++j) acc[i][j] = vzero8();

  for (int k0 = 0; k0 < K; k0 += 32) {
    v16bf a[4], b[4];
#pragma unroll
    for (int i = 0; i < 4; ++i) {
      if (ABF16) {
        const bf16_t* p = (const bf16_t*)Ap + (size_t)(m0 + i*16 + lo) * K + k0 + hi*8;
        cp8(a[i], 0, p); cp8(a[i], 8, p + 16);
      } else {
        const float* p = (const float*)Ap + (size_t)(m0 + i*16 + lo) * K + k0 + hi*8;
        cvt8(a[i], 0, p); cvt8(a[i], 8, p + 16);
      }
    }
#pragma unroll
    for (int j = 0; j < 4; ++j) {
      const float* p = W + (size_t)(n0 + j*16 + lo) * K + k0 + hi*8;
      cvt8(b[j], 0, p); cvt8(b[j], 8, p + 16);
    }
#pragma unroll
    for (int i = 0; i < 4; ++i)
#pragma unroll
      for (int j = 0; j < 4; ++j)
        acc[i][j] = wmma_bf16(a[i], b[j], acc[i][j]);
  }

#pragma unroll
  for (int i = 0; i < 4; ++i)
#pragma unroll
    for (int j = 0; j < 4; ++j)
#pragma unroll
      for (int r = 0; r < 8; ++r) {
        int row = m0 + i*16 + r + hi*8;
        int col = n0 + j*16 + lo;
        float v = acc[i][j][r];
        if (mode == 0) {
          outF[(size_t)row * N + col] = v;
        } else if (mode == 1) {
          outB[(size_t)row * N + col] = (bf16_t)v;
        } else {
          int kvh = col >> 7, d = col & 127;
          int bb = row >> 11, s = row & (SEQ_C - 1);
          outB[((size_t)(bb * NUM_KV_C + kvh) * HEAD_DIM_C + d) * SEQ_C + s] = (bf16_t)v;
        }
      }
}

// ---------------------------------------------------------------------------
// RoPE in place on bf16 Q ([T,4096]) and K ([T,1024]).
// pair (i, i+64) per head; inv_freq = 10000^(-i/64)
// ---------------------------------------------------------------------------
__global__ void rope_k(bf16_t* __restrict__ Qb, bf16_t* __restrict__ Kb,
                       const int* __restrict__ pos_ids) {
  int idx = blockIdx.x * blockDim.x + threadIdx.x;
  int i   = idx & 63;
  int rem = idx >> 6;
  int hh  = rem % (NUM_HEADS_C + NUM_KV_C);
  int t   = rem / (NUM_HEADS_C + NUM_KV_C);
  if (t >= TOKENS_C) return;
  float pos = (float)pos_ids[t];
  // ln(10000) = 9.210340371976184
  float inv = __expf(-(float)i * (9.210340371976184f / 64.f));
  float ang = pos * inv;
  float c = cosf(ang), s = sinf(ang);
  bf16_t* base;
  if (hh < NUM_HEADS_C) base = Qb + (size_t)t * HIDDEN_C + hh * HEAD_DIM_C;
  else                  base = Kb + (size_t)t * KVDIM_C + (hh - NUM_HEADS_C) * HEAD_DIM_C;
  float x1 = (float)base[i], x2 = (float)base[i + 64];
  base[i]      = (bf16_t)(x1 * c - x2 * s);
  base[i + 64] = (bf16_t)(x2 * c + x1 * s);
}

// ---------------------------------------------------------------------------
// Flash attention, causal, GQA (4 Q heads per KV head).
// One wave (blockDim=32) per 32 query rows of one (b, h).
// K and Vt tiles are staged into LDS with gfx1250 async global->LDS copies
// (fallback: load + ds_store); WMMA B-fragments are then fed from LDS.
// Q: [T,4096] bf16 (RoPE'd), K: [T,1024] bf16 (RoPE'd),
// Vt: [B,KVH,128,SEQ] bf16, O: [T,4096] bf16.
// ---------------------------------------------------------------------------
__global__ __launch_bounds__(32) void flash_k(const bf16_t* __restrict__ Q,
                                              const bf16_t* __restrict__ K,
                                              const bf16_t* __restrict__ Vt,
                                              bf16_t* __restrict__ O) {
  __shared__ bf16_t kls[32][136];  // K tile: 32 kv rows x 128 d (+pad, 272B stride)
  __shared__ bf16_t vls[128][40];  // V tile: 128 d rows x 32 kv (+pad, 80B stride)
  __shared__ bf16_t pls[32][40];   // P tile staging (C-layout -> A-layout)

  const int lane = threadIdx.x;
  const int lo = lane & 15, hi = lane >> 4;
  const int bh = blockIdx.x;
  const int b = bh / NUM_HEADS_C, hq = bh % NUM_HEADS_C;
  const int kvh = hq >> 2;                       // n_rep = 4
  const int q0 = blockIdx.y * 32;
  const float scale = 0.08838834764831845f;      // 1/sqrt(128)

  // Q fragments: 32 rows x 128 cols -> [mi][kk] (mi: M/16, kk: K/32)
  const bf16_t* qbase = Q + (size_t)(b * SEQ_C + q0) * HIDDEN_C + hq * HEAD_DIM_C;
  v16bf qf[2][4];
#pragma unroll
  for (int mi = 0; mi < 2; ++mi)
#pragma unroll
    for (int kk = 0; kk < 4; ++kk) {
      const bf16_t* p = qbase + (size_t)(mi*16 + lo) * HIDDEN_C + kk*32 + hi*8;
      cp8(qf[mi][kk], 0, p); cp8(qf[mi][kk], 8, p + 16);
    }

  v8f o[2][8];
  float mrun[2][8], lrun[2][8];
#pragma unroll
  for (int mi = 0; mi < 2; ++mi) {
#pragma unroll
    for (int nd = 0; nd < 8; ++nd) o[mi][nd] = vzero8();
#pragma unroll
    for (int r = 0; r < 8; ++r) { mrun[mi][r] = -1e30f; lrun[mi][r] = 0.f; }
  }

  const bf16_t* kbase  = K + (size_t)(b * SEQ_C) * KVDIM_C + kvh * HEAD_DIM_C;
  const bf16_t* vtbase = Vt + (size_t)(b * NUM_KV_C + kvh) * HEAD_DIM_C * SEQ_C;

  const int ntiles = q0 / 32 + 1;   // causal: kv <= q
  for (int kt = 0; kt < ntiles; ++kt) {
    const int kv0 = kt * 32;

    // ---- stage K tile (32 rows x 256B) : lane = kv row, 16 chunks of 16B
    {
      const bf16_t* krow = kbase + (size_t)(kv0 + lane) * KVDIM_C;
#pragma unroll
      for (int i = 0; i < 16; ++i)
        async_cp16(&kls[lane][i * 8], krow + i * 8);
      // ---- stage V tile (128 d rows x 64B) : item = d*4 + chunk
#pragma unroll
      for (int i = 0; i < 16; ++i) {
        int item = i * 32 + lane;
        int d = item >> 2, ch = item & 3;
        async_cp16(&vls[d][ch * 8], vtbase + (size_t)d * SEQ_C + kv0 + ch * 8);
      }
      async_join();
      __syncthreads();
    }

    // ---- S = Q K^T over this 32x32 tile (B-fragments from LDS)
    v8f sf[2][2];
#pragma unroll
    for (int mi = 0; mi < 2; ++mi)
#pragma unroll
      for (int ni = 0; ni < 2; ++ni) sf[mi][ni] = vzero8();

#pragma unroll
    for (int kk = 0; kk < 4; ++kk) {
      v16bf kb[2];
#pragma unroll
      for (int ni = 0; ni < 2; ++ni) {
        const bf16_t* p = &kls[ni*16 + lo][kk*32 + hi*8];
        cp8(kb[ni], 0, p); cp8(kb[ni], 8, p + 16);
      }
#pragma unroll
      for (int mi = 0; mi < 2; ++mi)
#pragma unroll
        for (int ni = 0; ni < 2; ++ni)
          sf[mi][ni] = wmma_bf16(qf[mi][kk], kb[ni], sf[mi][ni]);
    }

    // ---- online softmax + stage P (bf16) to LDS
#pragma unroll
    for (int mi = 0; mi < 2; ++mi) {
#pragma unroll
      for (int r = 0; r < 8; ++r) {
        const int qrow = q0 + mi*16 + r + hi*8;
        float v0 = sf[mi][0][r] * scale;
        float v1 = sf[mi][1][r] * scale;
        if (kv0 + lo      > qrow) v0 = -1e30f;
        if (kv0 + 16 + lo > qrow) v1 = -1e30f;

        float mx = fmaxf(v0, v1);
#pragma unroll
        for (int msk = 8; msk >= 1; msk >>= 1) mx = fmaxf(mx, __shfl_xor(mx, msk, 32));
        float nm = fmaxf(mrun[mi][r], mx);
        float alpha = __expf(mrun[mi][r] - nm);
        float p0 = __expf(v0 - nm);
        float p1 = __expf(v1 - nm);
        float rs = p0 + p1;
#pragma unroll
        for (int msk = 8; msk >= 1; msk >>= 1) rs += __shfl_xor(rs, msk, 32);
        lrun[mi][r] = lrun[mi][r] * alpha + rs;
        mrun[mi][r] = nm;
#pragma unroll
        for (int nd = 0; nd < 8; ++nd) o[mi][nd][r] *= alpha;

        const int rl = mi*16 + r + hi*8;
        pls[rl][lo]      = (bf16_t)p0;
        pls[rl][16 + lo] = (bf16_t)p1;
      }
    }
    __syncthreads();   // single-wave WG: LDS ops in-order; barrier for safety

    // ---- O += P @ V : P as A-frag (16x32), V B-frags from LDS
    v16bf pa[2];
#pragma unroll
    for (int mi = 0; mi < 2; ++mi) {
      const bf16_t* p = &pls[mi*16 + lo][hi*8];
      cp8(pa[mi], 0, p); cp8(pa[mi], 8, p + 16);
    }
#pragma unroll
    for (int nd = 0; nd < 8; ++nd) {
      v16bf vb;
      const bf16_t* p = &vls[nd*16 + lo][hi*8];
      cp8(vb, 0, p); cp8(vb, 8, p + 16);
#pragma unroll
      for (int mi = 0; mi < 2; ++mi)
        o[mi][nd] = wmma_bf16(pa[mi], vb, o[mi][nd]);
    }
    __syncthreads();   // protect kls/vls/pls before next iteration's staging
  }

  // normalize and write bf16 output, token-major [T, 4096]
  bf16_t* ob = O + (size_t)(b * SEQ_C + q0) * HIDDEN_C + hq * HEAD_DIM_C;
#pragma unroll
  for (int mi = 0; mi < 2; ++mi)
#pragma unroll
    for (int r = 0; r < 8; ++r) {
      float invl = 1.f / lrun[mi][r];
#pragma unroll
      for (int nd = 0; nd < 8; ++nd)
        ob[(size_t)(mi*16 + r + hi*8) * HIDDEN_C + nd*16 + lo] =
            (bf16_t)(o[mi][nd][r] * invl);
    }
}

// ---------------------------------------------------------------------------
// Launch
// ---------------------------------------------------------------------------
extern "C" void kernel_launch(void* const* d_in, const int* in_sizes, int n_in,
                              void* d_out, int out_size, void* d_ws, size_t ws_size,
                              hipStream_t stream) {
  const float* hs  = (const float*)d_in[0];
  const int*   pos = (const int*)  d_in[1];
  const float* wq  = (const float*)d_in[2];
  const float* wk  = (const float*)d_in[3];
  const float* wv  = (const float*)d_in[4];
  const float* wo  = (const float*)d_in[5];
  float* out = (float*)d_out;

  // workspace: Q(bf16 T*4096) + K(bf16 T*1024) + Vt(bf16 T*1024) + O(bf16 T*4096)
  bf16_t* Qb = (bf16_t*)d_ws;
  bf16_t* Kb = Qb + (size_t)TOKENS_C * HIDDEN_C;
  bf16_t* Vt = Kb + (size_t)TOKENS_C * KVDIM_C;
  bf16_t* Ob = Vt + (size_t)TOKENS_C * KVDIM_C;

  dim3 blk(256);
  // projections (A = fp32 hidden states)
  gemm_xwT<false><<<dim3(TOKENS_C/128, HIDDEN_C/256), blk, 0, stream>>>(
      hs, wq, HIDDEN_C, HIDDEN_C, nullptr, Qb, 1);
  gemm_xwT<false><<<dim3(TOKENS_C/128, KVDIM_C/256), blk, 0, stream>>>(
      hs, wk, HIDDEN_C, KVDIM_C, nullptr, Kb, 1);
  gemm_xwT<false><<<dim3(TOKENS_C/128, KVDIM_C/256), blk, 0, stream>>>(
      hs, wv, HIDDEN_C, KVDIM_C, nullptr, Vt, 2);

  // RoPE on Q and K
  int rope_total = TOKENS_C * (NUM_HEADS_C + NUM_KV_C) * (HEAD_DIM_C / 2);
  rope_k<<<rope_total / 256, 256, 0, stream>>>(Qb, Kb, pos);

  // flash attention
  flash_k<<<dim3(BATCH_C * NUM_HEADS_C, SEQ_C / 32), dim3(32), 0, stream>>>(
      Qb, Kb, Vt, Ob);

  // output projection (A = bf16 attention output) -> fp32 d_out
  gemm_xwT<true><<<dim3(TOKENS_C/128, HIDDEN_C/256), blk, 0, stream>>>(
      Ob, wo, HIDDEN_C, HIDDEN_C, out, nullptr, 0);
}
